// NeRF_57973468561819
// MI455X (gfx1250) — matrix operand order repaired and checked by reading, not compiled, not run on hardware
//
#include <hip/hip_runtime.h>

typedef _Float16 half_t;
typedef __attribute__((ext_vector_type(16))) _Float16 v16h;
typedef __attribute__((ext_vector_type(8)))  _Float16 v8h;
typedef __attribute__((ext_vector_type(8)))  float    v8f;

#define NFEAT 99
#define NSMP  32
#define NRAYS 50000
#define ASTR  192           // activation row stride (halves): [0..63 hidden | 64..191 input(pad)]
#define NWAVES 8
#define PTS_PER_BLOCK 128   // 8 waves * 16 points = 4 rays

// Transposed weight tiles in LDS: WT[n][k] f16, row stride = Kpad (halves)
#define WT_TOTAL 46080

__device__ __forceinline__ void lds_fence() {
  // cross-lane LDS RAW: wait for this wave's DS stores + block compiler reordering
  asm volatile("s_wait_dscnt 0" ::: "memory");
}

// A fragment 16x32 f16 (ISA 7.12.2): lane<16 holds M=lane, K = kb+{0..7, 16..23};
// lane>=16 holds M=lane-16, K = kb+{8..15, 24..31}.
__device__ __forceinline__ v16h load_a(const half_t* base, int kbase, int lane) {
  const int m  = lane & 15;
  const int kg = (lane >> 4) << 3;            // 0 or 8
  const half_t* rp = base + m * ASTR + kbase + kg;
  v8h lo = *(const v8h*)(rp);
  v8h hi = *(const v8h*)(rp + 16);
  return __builtin_shufflevector(lo, hi, 0,1,2,3,4,5,6,7,8,9,10,11,12,13,14,15);
}

// B fragment 32x16 f16: lane holds column N = lane&15; lane<16 -> K kb+{0..7, 16..23},
// lane>=16 -> K kb+{8..15, 24..31} (matches A fragment K order).
__device__ __forceinline__ v16h load_b(const half_t* wt, int kstride, int ntile,
                                       int kbase, int lane) {
  const int n  = (ntile << 4) + (lane & 15);
  const int kg = (lane >> 4) << 3;            // 0 or 8
  const half_t* rp = wt + n * kstride + kbase + kg;
  v8h lo = *(const v8h*)(rp);
  v8h hi = *(const v8h*)(rp + 16);
  return __builtin_shufflevector(lo, hi, 0,1,2,3,4,5,6,7,8,9,10,11,12,13,14,15);
}

// out[16x64] = relu(A[16xK] * WT^T + bias), result stored f16 into act cols 0..63.
__device__ __forceinline__ void dense_layer(const half_t* __restrict__ asrc, int ksteps,
                                            const half_t* __restrict__ wt, int kstride,
                                            const float* __restrict__ bias,
                                            half_t* __restrict__ outp, int lane) {
  lds_fence();
  v8f c0 = {}; v8f c1 = {}; v8f c2 = {}; v8f c3 = {};
  for (int ks = 0; ks < ksteps; ++ks) {
    const int kb = ks * 32;
    v16h a  = load_a(asrc, kb, lane);
    v16h b0 = load_b(wt, kstride, 0, kb, lane);
    c0 = __builtin_amdgcn_wmma_f32_16x16x32_f16(false, a, false, b0, (short)0, c0, false, false);
    v16h b1 = load_b(wt, kstride, 1, kb, lane);
    c1 = __builtin_amdgcn_wmma_f32_16x16x32_f16(false, a, false, b1, (short)0, c1, false, false);
    v16h b2 = load_b(wt, kstride, 2, kb, lane);
    c2 = __builtin_amdgcn_wmma_f32_16x16x32_f16(false, a, false, b2, (short)0, c2, false, false);
    v16h b3 = load_b(wt, kstride, 3, kb, lane);
    c3 = __builtin_amdgcn_wmma_f32_16x16x32_f16(false, a, false, b3, (short)0, c3, false, false);
  }
  // C layout: elem r -> (M = r + 8*(lane>=16), N = lane&15) per 16x16 f32 C/D table
  const int n  = lane & 15;
  const int mb = (lane >> 4) << 3;
  const float bv0 = bias[n], bv1 = bias[16 + n], bv2 = bias[32 + n], bv3 = bias[48 + n];
#pragma unroll
  for (int r = 0; r < 8; ++r) {
    float v;
    v = c0[r] + bv0; outp[(mb + r) * ASTR + n     ] = (half_t)(v > 0.f ? v : 0.f);
    v = c1[r] + bv1; outp[(mb + r) * ASTR + n + 16] = (half_t)(v > 0.f ? v : 0.f);
    v = c2[r] + bv2; outp[(mb + r) * ASTR + n + 32] = (half_t)(v > 0.f ? v : 0.f);
    v = c3[r] + bv3; outp[(mb + r) * ASTR + n + 48] = (half_t)(v > 0.f ? v : 0.f);
  }
}

__global__ __launch_bounds__(256) void nerf_fused_kernel(
    const float* __restrict__ rays, const float* __restrict__ tvals,
    const float* __restrict__ w0, const float* __restrict__ w1, const float* __restrict__ w2,
    const float* __restrict__ w3, const float* __restrict__ w4, const float* __restrict__ w5,
    const float* __restrict__ w6, const float* __restrict__ w7, const float* __restrict__ w8,
    const float* __restrict__ b0, const float* __restrict__ b1, const float* __restrict__ b2,
    const float* __restrict__ b3, const float* __restrict__ b4, const float* __restrict__ b5,
    const float* __restrict__ b6, const float* __restrict__ b7, const float* __restrict__ b8,
    float* __restrict__ out) {
  __shared__ __align__(32) half_t s_wt[WT_TOTAL];
  __shared__ float  s_bias[520];
  __shared__ __align__(32) half_t s_act[NWAVES * 16 * ASTR];
  __shared__ float  s_pred[PTS_PER_BLOCK * 4];

  const int tid = threadIdx.x;

  // ---- stage weights (f16, transposed WT[n][k], K zero-padded to Kpad) ----
  for (int i = tid; i < WT_TOTAL; i += 256) s_wt[i] = (half_t)0.f;
  __syncthreads();
  {
    const float* Wg[9] = {w0, w1, w2, w3, w4, w5, w6, w7, w8};
    const int Kv[9] = {99, 64, 64, 64, 64, 163, 64, 64, 64};
    const int Nn[9] = {64, 64, 64, 64, 64, 64, 64, 64, 4};
    const int Kp[9] = {128, 64, 64, 64, 64, 192, 64, 64, 64};
    const int WO[9] = {0, 8192, 12288, 16384, 20480, 24576, 36864, 40960, 45056};
    for (int L = 0; L < 9; ++L) {
      const float* Wsrc = Wg[L];
      const int kv = Kv[L], nn = Nn[L], kp = Kp[L], wo = WO[L];
      for (int idx = tid; idx < kv * nn; idx += 256) {
        const int k = idx / nn, n = idx - k * nn;
        s_wt[wo + n * kp + k] = (half_t)Wsrc[idx];
      }
    }
    const float* Bg[8] = {b0, b1, b2, b3, b4, b5, b6, b7};
    if (tid < 64)
      for (int L = 0; L < 8; ++L) s_bias[L * 64 + tid] = Bg[L][tid];
    if (tid < 4) s_bias[512 + tid] = b8[tid];
  }
  __syncthreads();

  const int wave = tid >> 5;
  const int lane = tid & 31;
  half_t* act = s_act + wave * 16 * ASTR;
  const long p0 = (long)blockIdx.x * PTS_PER_BLOCK + wave * 16;

  // ---- input features -> act cols 64..191 (99 valid, rest zero) ----
  for (int idx = lane; idx < 16 * 128; idx += 32) {
    const int r = idx >> 7, c = idx & 127;
    const float v = (c < NFEAT) ? rays[(p0 + r) * NFEAT + c] : 0.f;
    act[r * ASTR + 64 + c] = (half_t)v;
  }

  // ---- MLP: 8 hidden layers (skip concat is free by layout), WMMA f16 ----
  dense_layer(act + 64, 4, s_wt + 0,     128, s_bias + 0,   act, lane);  // L0: K=99->128
  dense_layer(act,      2, s_wt + 8192,  64,  s_bias + 64,  act, lane);  // L1
  dense_layer(act,      2, s_wt + 12288, 64,  s_bias + 128, act, lane);  // L2
  dense_layer(act,      2, s_wt + 16384, 64,  s_bias + 192, act, lane);  // L3
  dense_layer(act,      2, s_wt + 20480, 64,  s_bias + 256, act, lane);  // L4
  dense_layer(act,      6, s_wt + 24576, 192, s_bias + 320, act, lane);  // L5: K=163->192 (skip)
  dense_layer(act,      2, s_wt + 36864, 64,  s_bias + 384, act, lane);  // L6
  dense_layer(act,      2, s_wt + 40960, 64,  s_bias + 448, act, lane);  // L7

  // ---- final Dense(4): one N-tile, cols 0..3 valid ----
  lds_fence();
  v8f cf = {};
  for (int ks = 0; ks < 2; ++ks) {
    const int kb = ks * 32;
    v16h a = load_a(act, kb, lane);
    v16h b = load_b(s_wt + 45056, 64, 0, kb, lane);
    cf = __builtin_amdgcn_wmma_f32_16x16x32_f16(false, a, false, b, (short)0, cf, false, false);
  }
  {
    const int n = lane & 15, mb = (lane >> 4) << 3;
    if (n < 4) {
      const float bv = s_bias[512 + n];
#pragma unroll
      for (int r = 0; r < 8; ++r)
        s_pred[(wave * 16 + mb + r) * 4 + n] = cf[r] + bv;
    }
  }
  __syncthreads();

  // ---- fused volume rendering: 4 rays per block, one lane each ----
  if (tid < 4) {
    const long ray = (long)blockIdx.x * 4 + tid;
    const float* tv = tvals + ray * NSMP;
    float trans = 1.f, ar = 0.f, ag = 0.f, ab = 0.f, ad = 0.f;
    float tc = tv[0];
    for (int s = 0; s < NSMP; ++s) {
      const float tn = (s < NSMP - 1) ? tv[s + 1] : tc;
      const float delta = (s < NSMP - 1) ? (tn - tc) : 1e10f;
      const float* pr = s_pred + (tid * NSMP + s) * 4;
      float sg = pr[3];
      sg = sg > 0.f ? sg : 0.f;
      const float alpha = 1.f - __expf(-sg * delta);
      const float w = alpha * trans;
      ar += w / (1.f + __expf(-pr[0]));
      ag += w / (1.f + __expf(-pr[1]));
      ab += w / (1.f + __expf(-pr[2]));
      ad += w * tc;
      trans *= (1.f - alpha + 1e-10f);
      tc = tn;
    }
    out[ray * 3 + 0] = ar;
    out[ray * 3 + 1] = ag;
    out[ray * 3 + 2] = ab;
    out[(long)NRAYS * 3 + ray] = ad;
  }
}

extern "C" void kernel_launch(void* const* d_in, const int* in_sizes, int n_in,
                              void* d_out, int out_size, void* d_ws, size_t ws_size,
                              hipStream_t stream) {
  // setup_inputs() order: rays_flat, t_vals, Ws[0..8], bs[0..8]  (n_in == 20)
  const float* rays  = (const float*)d_in[0];
  const float* tvals = (const float*)d_in[1];
  const float* W[9];
  const float* Bb[9];
  for (int i = 0; i < 9; ++i) {
    W[i]  = (const float*)d_in[2 + i];
    Bb[i] = (const float*)d_in[11 + i];
  }
  float* out = (float*)d_out;

  const int npts   = in_sizes[0] / NFEAT;        // 1,600,000
  const int blocks = npts / PTS_PER_BLOCK;       // 12,500

  nerf_fused_kernel<<<blocks, 256, 0, stream>>>(
      rays, tvals,
      W[0], W[1], W[2], W[3], W[4], W[5], W[6], W[7], W[8],
      Bb[0], Bb[1], Bb[2], Bb[3], Bb[4], Bb[5], Bb[6], Bb[7], Bb[8],
      out);
}
